// MultiHeadAttention_11922829214488
// MI455X (gfx1250) — compile-verified, matrix operand order checked by reading, not run on hardware
//
#include <hip/hip_runtime.h>
#include <hip/hip_bf16.h>

typedef _Float16 f16;
typedef __attribute__((ext_vector_type(16))) _Float16 v16h;
typedef __attribute__((ext_vector_type(8)))  _Float16 v8h;
typedef __attribute__((ext_vector_type(8)))  float    v8f;
typedef __attribute__((ext_vector_type(4)))  unsigned v4u;

#define TT  4096
#define DD  768
#define NHH 12
#define HDD 64

// ---------------------------------------------------------------------------
// WMMA helpers (CDNA5 wave32, 16x16x32 f16 -> f32)
// ---------------------------------------------------------------------------
__device__ __forceinline__ v8f wmma_f16(v16h a, v16h b, v8f c) {
  return __builtin_amdgcn_wmma_f32_16x16x32_f16(
      /*neg_a=*/false, a, /*neg_b=*/false, b,
      /*c_mod=*/(short)0, c, /*reuse_a=*/false, /*reuse_b=*/false);
}

__device__ __forceinline__ v8f vzero8() {
  v8f z;
#pragma unroll
  for (int i = 0; i < 8; ++i) z[i] = 0.0f;
  return z;
}

// Load a 16x32 f16 A- or B-fragment from a row-major matrix (ld in elements).
// ISA layout: lane L -> row (L&15); half=L>>4 selects k-chunks [half*8 .. +7]
// and [16+half*8 .. +7]  => two 16-byte loads per lane.
__device__ __forceinline__ v16h load_ab_frag(const f16* base, int ld) {
  const int lane = threadIdx.x & 31;
  const int m    = lane & 15;
  const int hf   = lane >> 4;
  const f16* p = base + (size_t)m * (size_t)ld + hf * 8;
  v8h lo = *(const v8h*)(p);
  v8h hi = *(const v8h*)(p + 16);
  v16h r;
#pragma unroll
  for (int i = 0; i < 8; ++i) { r[i] = lo[i]; r[8 + i] = hi[i]; }
  return r;
}

// Issue one asynchronous 16-byte global->LDS copy (CDNA5 async DMA path,
// tracked by ASYNCcnt). lds_off is the byte offset inside the workgroup's LDS
// allocation (low 32 bits of a generic pointer to a __shared__ object).
__device__ __forceinline__ void async_copy_b128(unsigned lds_off,
                                                const void* gaddr) {
  asm volatile("global_load_async_to_lds_b128 %0, %1, off"
               :: "v"(lds_off), "v"(gaddr)
               : "memory");
}

__device__ __forceinline__ void wait_async_zero() {
  asm volatile("s_wait_asynccnt 0x0" ::: "memory");
}

// ---------------------------------------------------------------------------
// Conversion kernels
// ---------------------------------------------------------------------------
__global__ void cvt_f32_to_f16_kernel(const float* __restrict__ in,
                                      f16* __restrict__ out, int n) {
  int i = blockIdx.x * blockDim.x + threadIdx.x;
  if (i < n) out[i] = (f16)in[i];
}

// WT[n][k] = W[k][n]  (f32 -> f16 transpose, DD x DD)
__global__ void cvt_transpose_w_kernel(const float* __restrict__ W,
                                       f16* __restrict__ WT) {
  int i = blockIdx.x * blockDim.x + threadIdx.x;
  if (i >= DD * DD) return;
  int k = i / DD;
  int n = i % DD;
  WT[(size_t)n * DD + k] = (f16)W[i];
}

// Bit-pack mask into transposed bit plane: maskT[chunk][row], where bit j of
// maskT[c][r] == (mask[r][c*32 + j] != 0).  67 MB ints -> 2 MB bits.
__global__ void pack_mask_kernel(const int* __restrict__ mask,
                                 unsigned* __restrict__ maskT) {
  const int nchunk = TT / 32;
  int i = blockIdx.x * blockDim.x + threadIdx.x;   // i = row * nchunk + chunk
  if (i >= TT * nchunk) return;
  const int row   = i / nchunk;
  const int chunk = i % nchunk;
  const int* p = mask + (size_t)row * TT + chunk * 32;
  unsigned w = 0;
#pragma unroll
  for (int j = 0; j < 8; ++j) {
    v4u q = *(const v4u*)(p + j * 4);
    w |= (q.x != 0u ? 1u : 0u) << (j * 4 + 0);
    w |= (q.y != 0u ? 1u : 0u) << (j * 4 + 1);
    w |= (q.z != 0u ? 1u : 0u) << (j * 4 + 2);
    w |= (q.w != 0u ? 1u : 0u) << (j * 4 + 3);
  }
  maskT[(size_t)chunk * TT + row] = w;
}

// ---------------------------------------------------------------------------
// GEMM: Y[M][N] = A[M][K] (f16, row-major) * BT[N][K]^T (f16) + bias
// mode 0: store f16 row-major     (Q, K)
// mode 1: store f16 col-major Y^T (V -> VT[c][t], contiguous v8h stores)
// mode 2: store f32 row-major     (final output)
// One wave computes a 32x32 tile (2x2 WMMA frags, 4 WMMAs per k-step).
// ---------------------------------------------------------------------------
__global__ void __launch_bounds__(128)
gemm_wmma_kernel(const f16* __restrict__ A, const f16* __restrict__ BT,
                 const float* __restrict__ bias, void* __restrict__ outp,
                 int M, int N, int K, int mode) {
  const int wid = (blockIdx.x * blockDim.x + threadIdx.x) >> 5;
  const int nt  = N >> 5;
  const int tm  = wid / nt;
  const int tn  = wid % nt;
  if (tm * 32 >= M) return;

  const int lane = threadIdx.x & 31;
  const int n0   = lane & 15;
  const int hf   = lane >> 4;

  v8f acc00 = vzero8(), acc01 = vzero8(), acc10 = vzero8(), acc11 = vzero8();

  const f16* a0p = A + (size_t)(tm * 32 + 0)  * K;
  const f16* a1p = A + (size_t)(tm * 32 + 16) * K;
  const f16* b0p = BT + (size_t)(tn * 32 + 0)  * K;
  const f16* b1p = BT + (size_t)(tn * 32 + 16) * K;

  for (int k0 = 0; k0 < K; k0 += 32) {
    v16h a0 = load_ab_frag(a0p + k0, K);
    v16h a1 = load_ab_frag(a1p + k0, K);
    v16h b0 = load_ab_frag(b0p + k0, K);
    v16h b1 = load_ab_frag(b1p + k0, K);
    acc00 = wmma_f16(a0, b0, acc00);
    acc01 = wmma_f16(a0, b1, acc01);
    acc10 = wmma_f16(a1, b0, acc10);
    acc11 = wmma_f16(a1, b1, acc11);
  }

  v8f accs[2][2] = {{acc00, acc01}, {acc10, acc11}};
#pragma unroll
  for (int mi = 0; mi < 2; ++mi) {
#pragma unroll
    for (int nj = 0; nj < 2; ++nj) {
      const int col  = tn * 32 + nj * 16 + n0;
      const int rowb = tm * 32 + mi * 16 + hf * 8;  // 8 contiguous rows
      const float bb = bias[col];
      v8f v = accs[mi][nj];
      if (mode == 2) {
        float* O = (float*)outp;
#pragma unroll
        for (int r = 0; r < 8; ++r)
          O[(size_t)(rowb + r) * N + col] = v[r] + bb;
      } else if (mode == 0) {
        f16* O = (f16*)outp;
#pragma unroll
        for (int r = 0; r < 8; ++r)
          O[(size_t)(rowb + r) * N + col] = (f16)(v[r] + bb);
      } else {  // mode 1: transposed store, contiguous in M
        f16* O = (f16*)outp;
        v8h st;
#pragma unroll
        for (int r = 0; r < 8; ++r) st[r] = (f16)(v[r] + bb);
        *(v8h*)(O + (size_t)col * M + rowb) = st;
      }
    }
  }
}

// ---------------------------------------------------------------------------
// Flash attention. One block = 4 waves = 64 query rows of ONE head, so the
// K-chunk (32x64) and V-chunk (64x32) are staged ONCE per block into LDS with
// double-buffered GLOBAL_LOAD_ASYNC_TO_LDS_B128 copies (ASYNCcnt), overlapping
// the DMA of chunk i+1 with WMMA compute on chunk i.
// Q,K row-major f16 [TT][DD]; VT f16 [DD][TT] (VT[c][t] = V[t][c]).
// maskT bit plane [TT/32][TT]: bit j of maskT[c][r] set -> score(r, c*32+j)
// replaced by 1e-20 (reference masked_fill; no -inf, no 1/sqrt(d) scaling).
// Online softmax in f32; row sums computed as P @ ones via WMMA.
// ---------------------------------------------------------------------------
__global__ void __launch_bounds__(128)
flash_attn_wmma_kernel(const f16* __restrict__ Q, const f16* __restrict__ Km,
                       const f16* __restrict__ VT,
                       const unsigned* __restrict__ maskT,
                       f16* __restrict__ O16) {
  __shared__ __align__(16) f16 Kbuf[2][32][64];   // [buf][key row][head dim]
  __shared__ __align__(16) f16 Vbuf[2][64][32];   // [buf][head dim][key]
  __shared__ __align__(16) f16 plds[4][16][32];   // per-wave P staging

  const int tid   = threadIdx.x;
  const int wslot = tid >> 5;
  const int h     = blockIdx.x / (TT / 64);
  const int qg    = blockIdx.x % (TT / 64);
  const int qbase = qg * 64 + wslot * 16;

  const int lane = tid & 31;
  const int n0   = lane & 15;
  const int hf   = lane >> 4;

  // Issue async copies for one 32-key chunk: K 32x64 and V 64x32 halves,
  // each 256 x 16B transfers; 128 threads issue 2+2 copies apiece.
  auto stage = [&](int buf, int kb) {
#pragma unroll
    for (int i = 0; i < 2; ++i) {
      const int c  = tid + i * 128;
      const int kr = c >> 3, kc = (c & 7) * 8;
      async_copy_b128((unsigned)(size_t)(void*)&Kbuf[buf][kr][kc],
                      Km + (size_t)(kb + kr) * DD + h * HDD + kc);
      const int vr = c >> 2, vc = (c & 3) * 8;
      async_copy_b128((unsigned)(size_t)(void*)&Vbuf[buf][vr][vc],
                      VT + (size_t)(h * HDD + vr) * TT + kb + vc);
    }
  };

  // Q fragments for this wave's tile (k = 0..63 of this head), resident.
  const f16* qp = Q + (size_t)qbase * DD + h * HDD;
  v16h aq0 = load_ab_frag(qp, DD);
  v16h aq1 = load_ab_frag(qp + 32, DD);

  v16h ones;
#pragma unroll
  for (int i = 0; i < 16; ++i) ones[i] = (f16)1.0f;

  v8f o0 = vzero8(), o1 = vzero8(), o2 = vzero8(), o3 = vzero8();
  float mrow[8], lrow[8];
#pragma unroll
  for (int r = 0; r < 8; ++r) { mrow[r] = -3.0e38f; lrow[r] = 0.0f; }

  stage(0, 0);
  int buf = 0;
  for (int kb = 0; kb < TT; kb += 32) {
    wait_async_zero();     // this wave's share of the staged chunk landed
    __syncthreads();       // all waves' shares landed -> buf is ready
    if (kb + 32 < TT) stage(buf ^ 1, kb + 32);   // overlap next chunk DMA

    // ---- S = Q * K^T for 32 keys (two 16-col C frags), K from LDS ----
    v16h bk00 = load_ab_frag(&Kbuf[buf][0][0],   64);
    v16h bk01 = load_ab_frag(&Kbuf[buf][0][32],  64);
    v16h bk10 = load_ab_frag(&Kbuf[buf][16][0],  64);
    v16h bk11 = load_ab_frag(&Kbuf[buf][16][32], 64);
    v8f s0 = vzero8(), s1 = vzero8();
    s0 = wmma_f16(aq0, bk00, s0);
    s0 = wmma_f16(aq1, bk01, s0);
    s1 = wmma_f16(aq0, bk10, s1);
    s1 = wmma_f16(aq1, bk11, s1);

    // ---- mask bits: this lane owns rows qbase + hf*8 + (0..7) ----
    unsigned mw[8];
    {
      const unsigned* mwp = maskT + (size_t)(kb >> 5) * TT + qbase + hf * 8;
      *(v4u*)(&mw[0]) = *(const v4u*)(mwp);
      *(v4u*)(&mw[4]) = *(const v4u*)(mwp + 4);
    }
#pragma unroll
    for (int r = 0; r < 8; ++r) {
      if ((mw[r] >> n0) & 1u)        s0[r] = 1e-20f;
      if ((mw[r] >> (16 + n0)) & 1u) s1[r] = 1e-20f;
    }

    // ---- online softmax: row max across the 16 lanes of this half ----
    float cm[8];
#pragma unroll
    for (int r = 0; r < 8; ++r) cm[r] = fmaxf(s0[r], s1[r]);
#pragma unroll
    for (int off = 8; off >= 1; off >>= 1)
#pragma unroll
      for (int r = 0; r < 8; ++r)
        cm[r] = fmaxf(cm[r], __shfl_xor(cm[r], off, 32));

    float scale[8];
#pragma unroll
    for (int r = 0; r < 8; ++r) {
      const float mn = fmaxf(mrow[r], cm[r]);
      scale[r] = __expf(mrow[r] - mn);
      mrow[r]  = mn;
      s0[r] = __expf(s0[r] - mn);
      s1[r] = __expf(s1[r] - mn);
    }

    // rescale O accumulators
#pragma unroll
    for (int r = 0; r < 8; ++r) {
      o0[r] *= scale[r]; o1[r] *= scale[r];
      o2[r] *= scale[r]; o3[r] *= scale[r];
    }

    // ---- P (C layout) -> LDS -> A-fragment layout ----
#pragma unroll
    for (int r = 0; r < 8; ++r) {
      const int m = r + hf * 8;
      plds[wslot][m][n0]      = (f16)s0[r];
      plds[wslot][m][16 + n0] = (f16)s1[r];
    }
    v16h ap = load_ab_frag(&plds[wslot][0][0], 32);

    // ---- row sums via WMMA: rs = P @ ones lands exactly on lrow[r] ----
    v8f rsf = wmma_f16(ap, ones, vzero8());
#pragma unroll
    for (int r = 0; r < 8; ++r) lrow[r] = lrow[r] * scale[r] + rsf[r];

    // ---- O += P * V, V from LDS (transposed chunk Vbuf[dim][key]) ----
    v16h bv0 = load_ab_frag(&Vbuf[buf][0][0],  32);
    v16h bv1 = load_ab_frag(&Vbuf[buf][16][0], 32);
    v16h bv2 = load_ab_frag(&Vbuf[buf][32][0], 32);
    v16h bv3 = load_ab_frag(&Vbuf[buf][48][0], 32);
    o0 = wmma_f16(ap, bv0, o0);
    o1 = wmma_f16(ap, bv1, o1);
    o2 = wmma_f16(ap, bv2, o2);
    o3 = wmma_f16(ap, bv3, o3);

    __syncthreads();       // everyone done reading buf before it is reused
    buf ^= 1;
  }

  // ---- epilogue: divide by l, store f16 row-major ----
  v8f os[4] = {o0, o1, o2, o3};
#pragma unroll
  for (int jn = 0; jn < 4; ++jn) {
#pragma unroll
    for (int r = 0; r < 8; ++r) {
      const int row = qbase + r + hf * 8;
      const int col = h * HDD + jn * 16 + n0;
      O16[(size_t)row * DD + col] = (f16)(os[jn][r] / lrow[r]);
    }
  }
}

// ---------------------------------------------------------------------------
// Launch
// ---------------------------------------------------------------------------
extern "C" void kernel_launch(void* const* d_in, const int* in_sizes, int n_in,
                              void* d_out, int out_size, void* d_ws, size_t ws_size,
                              hipStream_t stream) {
  const float* x    = (const float*)d_in[0];
  const int*   mask = (const int*)d_in[1];
  const float* Wq   = (const float*)d_in[2];
  const float* bq   = (const float*)d_in[3];
  const float* Wk   = (const float*)d_in[4];
  const float* bk   = (const float*)d_in[5];
  const float* Wv   = (const float*)d_in[6];
  const float* bv   = (const float*)d_in[7];
  const float* Wo   = (const float*)d_in[8];
  const float* bo   = (const float*)d_in[9];
  float* out = (float*)d_out;

  char* ws = (char*)d_ws;
  size_t off = 0;
  auto alloc = [&](size_t bytes) -> char* {
    char* p = ws + off;
    off += (bytes + 255) & ~(size_t)255;
    return p;
  };
  const size_t TD = (size_t)TT * DD;
  f16* x16  = (f16*)alloc(TD * 2);
  f16* WqT  = (f16*)alloc((size_t)DD * DD * 2);
  f16* WkT  = (f16*)alloc((size_t)DD * DD * 2);
  f16* WvT  = (f16*)alloc((size_t)DD * DD * 2);
  f16* WoT  = (f16*)alloc((size_t)DD * DD * 2);
  f16* Q16  = (f16*)alloc(TD * 2);
  f16* K16  = (f16*)alloc(TD * 2);
  f16* VT16 = (f16*)alloc(TD * 2);   // [DD][TT]
  f16* O16  = (f16*)alloc(TD * 2);
  unsigned* maskT = (unsigned*)alloc((size_t)(TT / 32) * TT * 4);
  (void)ws_size; (void)n_in; (void)in_sizes; (void)out_size;

  // 1) precision conversion, weight transposes, mask bit-packing
  {
    int n = (int)TD;
    cvt_f32_to_f16_kernel<<<(n + 255) / 256, 256, 0, stream>>>(x, x16, n);
    int nw = DD * DD;
    dim3 g((nw + 255) / 256);
    cvt_transpose_w_kernel<<<g, 256, 0, stream>>>(Wq, WqT);
    cvt_transpose_w_kernel<<<g, 256, 0, stream>>>(Wk, WkT);
    cvt_transpose_w_kernel<<<g, 256, 0, stream>>>(Wv, WvT);
    cvt_transpose_w_kernel<<<g, 256, 0, stream>>>(Wo, WoT);
    int nm = TT * (TT / 32);
    pack_mask_kernel<<<(nm + 255) / 256, 256, 0, stream>>>(mask, maskT);
  }

  // 2) projections: (TT x DD) * (DD x DD); 3072 waves -> 768 blocks of 128
  const int gemm_waves  = (TT / 32) * (DD / 32);
  const int gemm_blocks = gemm_waves / 4;
  gemm_wmma_kernel<<<gemm_blocks, 128, 0, stream>>>(x16, WqT, bq, Q16,
                                                    TT, DD, DD, /*mode=*/0);
  gemm_wmma_kernel<<<gemm_blocks, 128, 0, stream>>>(x16, WkT, bk, K16,
                                                    TT, DD, DD, /*mode=*/0);
  gemm_wmma_kernel<<<gemm_blocks, 128, 0, stream>>>(x16, WvT, bv, VT16,
                                                    TT, DD, DD, /*mode=*/1);

  // 3) attention: one block (4 waves) per 64 query rows of one head
  const int fa_blocks = NHH * (TT / 64);   // 768
  flash_attn_wmma_kernel<<<fa_blocks, 128, 0, stream>>>(Q16, K16, VT16, maskT, O16);

  // 4) output projection -> f32
  gemm_wmma_kernel<<<gemm_blocks, 128, 0, stream>>>(O16, WoT, bo, out,
                                                    TT, DD, DD, /*mode=*/2);
}
